// GCN_31722628448490
// MI455X (gfx1250) — matrix operand order repaired and checked by reading, not compiled, not run on hardware
//
#include <hip/hip_runtime.h>
#include <hip/hip_bf16.h>

typedef __attribute__((ext_vector_type(16))) _Float16 v16h;
typedef __attribute__((ext_vector_type(8)))  _Float16 v8h;
typedef __attribute__((ext_vector_type(8)))  float    v8f;

#define J_ 29
#define C_ 3
#define JC 87            // J*C flattened features per frame
#define KTILES 3         // 3 * 32 = 96 >= 87 (K padded)
#define NTILES 6         // 6 * 16 = 96 >= 87 (N padded)
#define NFRAGS (NTILES * KTILES)   // 18
#define FRAG_ELEMS 512             // 32 lanes * 16 halfs per B fragment
#define WG_FRAMES 128
#define ROWPAD 104       // halfs per LDS row: 52 dwords -> 16B-aligned b128
                         // segments, conflict-free banking (13 coprime 16),
                         // covers K pads 87..103 so no guards in inner loop

// ------------------------------------------------------------------
// Prep: build Ahat = D^-1/2 (A+I) D^-1/2, fold with W into the 87x87
// map Mt[K][N] = Ahat[N/3][K/3] * W[(K%3)*3 + (N%3)], and store it as
// f16 WMMA B-fragments: frag(nt,kt) -> [lane][elem], N = nt*16+lane%16,
// K = kt*32 + 16*(lane/16) + elem  (contiguous k order, same as the A reads).
// ------------------------------------------------------------------
__global__ void gcn_prep(const long long* __restrict__ edge, int E,
                         const float* __restrict__ W,
                         _Float16* __restrict__ wsB) {
  __shared__ float Ahat[J_][J_];
  int tid = threadIdx.x;
  for (int i = tid; i < J_ * J_; i += blockDim.x) ((float*)Ahat)[i] = 0.f;
  __syncthreads();
  if (tid == 0) {
    float deg[J_], dinv[J_];
    for (int j = 0; j < J_; ++j) deg[j] = 0.f;
    for (int e = 0; e < E + J_; ++e) {
      int d = (e < E) ? (int)edge[E + e] : (e - E);   // self loops appended
      deg[d] += 1.f;
    }
    for (int j = 0; j < J_; ++j) dinv[j] = deg[j] > 0.f ? rsqrtf(deg[j]) : 0.f;
    for (int e = 0; e < E + J_; ++e) {
      int s = (e < E) ? (int)edge[e]     : (e - E);
      int d = (e < E) ? (int)edge[E + e] : (e - E);
      Ahat[d][s] += dinv[s] * dinv[d];
    }
  }
  __syncthreads();
  for (int idx = tid; idx < NFRAGS * FRAG_ELEMS; idx += blockDim.x) {
    int frag   = idx >> 9;       // /512
    int within = idx & 511;
    int lane = within >> 4;
    int e    = within & 15;
    int nt = frag / KTILES, kt = frag % KTILES;
    int N = nt * 16 + (lane & 15);
    int g = lane >> 4;
    int K = kt * 32 + 16 * g + e;
    float v = 0.f;
    if (N < JC && K < JC)
      v = Ahat[N / C_][K / C_] * W[(K % C_) * C_ + (N % C_)];
    wsB[idx] = (_Float16)v;
  }
}

__global__ void gcn_zero(float* __restrict__ out, int n) {
  int i = blockIdx.x * blockDim.x + threadIdx.x;
  if (i < n) out[i] = 0.f;
}

// ------------------------------------------------------------------
// Main streaming kernel: 256 threads = 8 waves; block stages 128 frames
// (contiguous, 16B-aligned) into LDS as f16 rows of 104; each wave does a
// 16-frame x 96-feature GEMM tile with v_wmma_f32_16x16x32_f16, then
// bias+ReLU+frame-reduction and atomic adds into the per-batch h sums.
// ------------------------------------------------------------------
__global__ void __launch_bounds__(256)
gcn_main(const float* __restrict__ x,
         const _Float16* __restrict__ wsB,
         const float* __restrict__ bvec,
         float* __restrict__ hout,
         int framesPerBatch) {
  __shared__ _Float16 xs[WG_FRAMES * ROWPAD];
  const int tid = threadIdx.x;

  // ---- stage: coalesced float4 global loads -> f16 LDS (rows padded) ----
  const float4* __restrict__ gx =
      (const float4*)(x + (long long)blockIdx.x * (WG_FRAMES * JC));
  const int nvec4 = (WG_FRAMES * JC) / 4;   // 2784, base offset 16B-aligned
  for (int i = tid; i < nvec4; i += blockDim.x) {
    float4 v = gx[i];
    int ff = i * 4;
    float vv[4] = {v.x, v.y, v.z, v.w};
#pragma unroll
    for (int c = 0; c < 4; ++c) {
      int f = (ff + c) / JC, col = (ff + c) % JC;
      xs[f * ROWPAD + col] = (_Float16)vv[c];
    }
  }
  // zero pad cols 87..103: col 87 as b16, cols 88..103 as 8 dwords per row
  if (tid < WG_FRAMES) xs[tid * ROWPAD + JC] = (_Float16)0.f;
  {
    unsigned* xsw = (unsigned*)xs;
    for (int i = tid; i < WG_FRAMES * 8; i += blockDim.x) {
      int row = i >> 3, d = i & 7;
      xsw[row * (ROWPAD / 2) + 44 + d] = 0u;
    }
  }
  __syncthreads();

  // ---- per-wave 16x96x96 WMMA tile ----
  const int wave = tid >> 5;
  const int lane = tid & 31;
  const int lrow = lane & 15;
  const int g    = lane >> 4;
  const int rowBase = wave * 16;

  v8f cacc[NTILES] = {};

#pragma unroll
  for (int kt = 0; kt < KTILES; ++kt) {
    // A fragment matching B packing: elem e -> K = kt*32 + 16g + e
    // = two 16B-aligned contiguous 8-half segments per lane.
    const int row = rowBase + lrow;
    const _Float16* rp = &xs[row * ROWPAD + kt * 32 + 16 * g];
    v8h lo = *(const v8h*)(rp);        // K = kt*32 + 16g + 0..7
    v8h hi = *(const v8h*)(rp + 8);    // K = kt*32 + 16g + 8..15
    v16h afrag = __builtin_shufflevector(lo, hi, 0, 1, 2, 3, 4, 5, 6, 7,
                                         8, 9, 10, 11, 12, 13, 14, 15);
#pragma unroll
    for (int nt = 0; nt < NTILES; ++nt) {
      v16h bfrag = *(const v16h*)(wsB + ((nt * KTILES + kt) * 32 + lane) * 16);
      cacc[nt] = __builtin_amdgcn_wmma_f32_16x16x32_f16(
          false, afrag, false, bfrag, (short)0, cacc[nt], false, false);
    }
  }

  // ---- bias + ReLU + reduce over 16 frames, atomic into h sums ----
  const float b0 = bvec[0], b1 = bvec[1], b2 = bvec[2];
  const int frame0 = blockIdx.x * WG_FRAMES + rowBase;
  const int batch  = frame0 / framesPerBatch;   // 16-frame tile never straddles
#pragma unroll
  for (int nt = 0; nt < NTILES; ++nt) {
    int N = nt * 16 + lrow;                 // C layout: lane%16 = N
    int m3 = N % 3;
    float bias = (m3 == 0) ? b0 : (m3 == 1) ? b1 : b2;
    float s = 0.f;
#pragma unroll
    for (int r = 0; r < 8; ++r) {           // VGPR r = frame row r + 8*g
      float v = cacc[nt][r] + bias;
      s += (v > 0.f) ? v : 0.f;
    }
    s += __shfl_xor(s, 16, 32);             // combine M groups (r vs r+8)
    if (lane < 16 && N < JC)
      atomicAdd(&hout[batch * JC + N], s);
  }
}

// ------------------------------------------------------------------
// Finalize: h = sum/num_frames (in place), z = sigmoid(h @ fcW + fcb)
// ------------------------------------------------------------------
__global__ void gcn_final(float* __restrict__ out,
                          const float* __restrict__ fcW,
                          const float* __restrict__ fcb,
                          int B, int framesPerBatch) {
  __shared__ float hv[JC];
  int b = blockIdx.x, tid = threadIdx.x;
  float inv = 1.0f / (float)framesPerBatch;
  if (tid < JC) {
    float v = out[b * JC + tid] * inv;
    out[b * JC + tid] = v;
    hv[tid] = v;
  }
  __syncthreads();
  if (tid < 2) {
    float acc = fcb[tid];
    for (int n = 0; n < JC; ++n) acc += hv[n] * fcW[n * 2 + tid];
    out[B * JC + b * 2 + tid] = 1.0f / (1.0f + expf(-acc));
  }
}

extern "C" void kernel_launch(void* const* d_in, const int* in_sizes, int n_in,
                              void* d_out, int out_size, void* d_ws, size_t ws_size,
                              hipStream_t stream) {
  (void)n_in; (void)ws_size;
  const float*     x    = (const float*)d_in[0];
  const long long* edge = (const long long*)d_in[1];
  const float*     W    = (const float*)d_in[4];
  const float*     b    = (const float*)d_in[5];
  const float*     fcW  = (const float*)d_in[6];
  const float*     fcb  = (const float*)d_in[7];
  float*     out = (float*)d_out;
  _Float16*  wsB = (_Float16*)d_ws;

  const int       E    = in_sizes[1] / 2;                 // 56 edges
  const long long ntot = (long long)in_sizes[0] / C_;     // total nodes
  const long long Ftot = ntot / J_;                       // total frames
  const int       B    = out_size / (JC + 2);             // 256 batches
  const int       framesPerBatch = (int)(Ftot / B);       // 1024

  gcn_prep<<<1, 128, 0, stream>>>(edge, E, W, wsB);
  const int hN = B * JC;
  gcn_zero<<<(hN + 255) / 256, 256, 0, stream>>>(out, hN);
  const int nblocks = (int)(Ftot / WG_FRAMES);            // 2048
  gcn_main<<<nblocks, 256, 0, stream>>>(x, wsB, b, out, framesPerBatch);
  gcn_final<<<B, 128, 0, stream>>>(out, fcW, fcb, B, framesPerBatch);
}
